// Block_6992206758417
// MI455X (gfx1250) — matrix operand order repaired and checked by reading, not compiled, not run on hardware
//
#include <hip/hip_runtime.h>
#include <hip/hip_bf16.h>

// ---------------------------------------------------------------------------
// MI455X (gfx1250) transformer block forward.
// bf16 WMMA (v_wmma_f32_16x16x32_bf16) for all GEMMs, fp32 softmax/LN/resid.
// Global->LDS staging uses CDNA5 async DMA (global_load_async_to_lds_b128,
// ASYNCcnt) with double-buffered K-loop in the GEMM.
// ---------------------------------------------------------------------------

typedef __bf16 bf16_t;
typedef __attribute__((ext_vector_type(16))) __bf16 v16bf;
typedef __attribute__((ext_vector_type(8)))  __bf16 v8bf;
typedef __attribute__((ext_vector_type(8)))  float  v8f;

#define WMMA_BF16(a, b, c) \
  __builtin_amdgcn_wmma_f32_16x16x32_bf16(false, (a), false, (b), (short)0, (c), false, false)

// CDNA5 async global->LDS DMA (GVS mode: 64-bit SGPR base + 32-bit VGPR byte
// offset). The LDS destination VGPR holds addr[31:0] of the generic pointer
// (ISA 10.2: LDS aperture maps low 32 bits to the LDS address).
#define ASYNC_LDS_B128(lds_addr, goff, base)                        \
  asm volatile("global_load_async_to_lds_b128 %0, %1, %2"           \
               :: "v"(lds_addr), "v"(goff), "s"(base) : "memory")
#define WAIT_ASYNC() asm volatile("s_wait_asynccnt 0x0" ::: "memory")

__device__ __forceinline__ uint32_t lds_lo32(const void* p) {
  return (uint32_t)(uintptr_t)p;
}

// --- fragment loaders (ISA 7.12.2 layouts; LDS pitches keep 16B alignment) ---

// A 16x32 bf16: lane m = lane%16 is the row; half h = lane/16.
// elems 0..7 -> K = h*8 + e ; elems 8..15 -> K = 16 + h*8 + e.
__device__ __forceinline__ v16bf ld_a_bf16(const bf16_t* base, int row0, int pitch,
                                           int k0, int lane) {
  const int m = lane & 15, h = lane >> 4;
  const bf16_t* p = base + (size_t)(row0 + m) * pitch + k0 + h * 8;
  v8bf lo = *(const v8bf*)p;
  v8bf hi = *(const v8bf*)(p + 16);
  v16bf r;
#pragma unroll
  for (int i = 0; i < 8; ++i) { r[i] = lo[i]; r[i + 8] = hi[i]; }
  return r;
}

// B 32x16 bf16 stored transposed in LDS (Bt[n][k]): lane n = lane%16 is the
// column; half h = lane/16: elems 0..7 -> K = h*8 + e ; elems 8..15 -> K = 16 + h*8 + e.
__device__ __forceinline__ v16bf ld_b_bf16(const bf16_t* base, int col0, int pitch,
                                           int k0, int lane) {
  const int n = lane & 15, h = lane >> 4;
  const bf16_t* p = base + (size_t)(col0 + n) * pitch + k0 + h * 8;
  v8bf lo = *(const v8bf*)p;
  v8bf hi = *(const v8bf*)(p + 16);
  v16bf r;
#pragma unroll
  for (int i = 0; i < 8; ++i) { r[i] = lo[i]; r[i + 8] = hi[i]; }
  return r;
}

// A fragment built from fp32 LDS (softmax probabilities), converted to bf16.
__device__ __forceinline__ v16bf ld_a_f32(const float* base, int row0, int pitch,
                                          int k0, int lane) {
  const int m = lane & 15, h = lane >> 4;
  const float* p = base + (size_t)(row0 + m) * pitch + k0;
  v16bf r;
#pragma unroll
  for (int i = 0; i < 8; ++i) r[i] = (bf16_t)p[h * 8 + i];
#pragma unroll
  for (int i = 0; i < 8; ++i) r[i + 8] = (bf16_t)p[16 + h * 8 + i];
  return r;
}

// ---------------------------------------------------------------------------
// Weight prep: fp32 -> bf16, transposed to Bt[N][K] so GEMM tile loads are
// contiguous along K.
// ---------------------------------------------------------------------------
__global__ __launch_bounds__(256) void qkvw_pack_kernel(
    const float* __restrict__ wq, const float* __restrict__ wk,
    const float* __restrict__ wv, bf16_t* __restrict__ wt) {
  int idx = blockIdx.x * 256 + threadIdx.x;      // wt[nrow][c], nrow in [0,1536)
  if (idx >= 3 * 512 * 512) return;
  int nrow = idx >> 9, c = idx & 511;
  int which = nrow >> 9, hh = (nrow >> 6) & 7, d = nrow & 63;
  const float* w = (which == 0) ? wq : (which == 1) ? wk : wv;   // [8][512][64]
  wt[idx] = (bf16_t)w[((size_t)hh * 512 + c) * 64 + d];
}

__global__ __launch_bounds__(256) void transpose_bf16_kernel(
    const float* __restrict__ w, bf16_t* __restrict__ wt, int K, int N) {
  int idx = blockIdx.x * 256 + threadIdx.x;      // wt[n][k] = w[k][n]
  if (idx >= N * K) return;
  int n = idx / K, k = idx - n * K;
  wt[idx] = (bf16_t)w[(size_t)k * N + n];
}

// ---------------------------------------------------------------------------
// LayerNorm over C=512, one 256-thread block (8 waves) per row. fp32 math,
// bf16 output for the downstream WMMA GEMM.
// ---------------------------------------------------------------------------
__global__ __launch_bounds__(256) void layernorm_bf16_kernel(
    const float* __restrict__ x, const float* __restrict__ g,
    const float* __restrict__ b, bf16_t* __restrict__ out) {
  __shared__ float rs_[8], rq_[8];
  const int row = blockIdx.x, tid = threadIdx.x;
  const float* xr = x + (size_t)row * 512;
  float2 v = *(const float2*)(xr + tid * 2);
  float s = v.x + v.y;
  float q = v.x * v.x + v.y * v.y;
#pragma unroll
  for (int o = 16; o >= 1; o >>= 1) {
    s += __shfl_xor(s, o, 32);
    q += __shfl_xor(q, o, 32);
  }
  if ((tid & 31) == 0) { rs_[tid >> 5] = s; rq_[tid >> 5] = q; }
  __syncthreads();
  float ts = 0.f, tq = 0.f;
#pragma unroll
  for (int i = 0; i < 8; ++i) { ts += rs_[i]; tq += rq_[i]; }
  const float mu = ts * (1.f / 512.f);
  const float var = tq * (1.f / 512.f) - mu * mu;
  const float inv = rsqrtf(var + 1e-5f);
  const int c = tid * 2;
  float2 gg = *(const float2*)(g + c);
  float2 bb = *(const float2*)(b + c);
  out[(size_t)row * 512 + c]     = (bf16_t)((v.x - mu) * inv * gg.x + bb.x);
  out[(size_t)row * 512 + c + 1] = (bf16_t)((v.y - mu) * inv * gg.y + bb.y);
}

// ---------------------------------------------------------------------------
// Generic bf16 WMMA GEMM: C[M,N] = A[M,K] * Bt[N,K]^T, fp32 accumulate.
// BM=64 BN=64 BK=64; 256 threads = 8 waves in a 2x4 grid; each wave owns a
// 32x16 output strip (2 accumulators, 4 WMMAs per K-step). Tiles are fetched
// with async global->LDS DMA, double buffered so DMA overlaps WMMA.
// MODE 0: scatter bf16 to q|k|v [B,H,T,D]      MODE 1: fp32 +bias+resid
// MODE 2: bf16 relu(+bias)                     MODE 3: fp32 +bias+resid (out)
// ---------------------------------------------------------------------------
template <int MODE>
__global__ __launch_bounds__(256) void gemm_bf16_kernel(
    const bf16_t* __restrict__ A, const bf16_t* __restrict__ Bt,
    int M, int N, int K,
    const float* __restrict__ bias, const float* __restrict__ resid,
    float* __restrict__ outF, bf16_t* __restrict__ outB) {
  __shared__ __align__(16) bf16_t As[2][64 * 72];   // pitch 72 elems = 144B
  __shared__ __align__(16) bf16_t Bs[2][64 * 72];
  const int tid = threadIdx.x, lane = tid & 31, wave = tid >> 5;
  const int wr = wave >> 2, wc = wave & 3;
  const int bx = blockIdx.x, by = blockIdx.y;

  // each thread DMAs 2 x 16B chunks per tile (64 rows x 64 k)
  const int ch0 = tid * 2, ch1 = ch0 + 1;
  const int r0 = ch0 >> 3, c0 = (ch0 & 7) * 8;
  const int r1 = ch1 >> 3, c1 = (ch1 & 7) * 8;

  auto issue_tile = [&](int buf, int k0) {
    const uint32_t ga0 = (uint32_t)((((size_t)(by * 64 + r0)) * K + k0 + c0) * 2);
    const uint32_t ga1 = (uint32_t)((((size_t)(by * 64 + r1)) * K + k0 + c1) * 2);
    const uint32_t gb0 = (uint32_t)((((size_t)(bx * 64 + r0)) * K + k0 + c0) * 2);
    const uint32_t gb1 = (uint32_t)((((size_t)(bx * 64 + r1)) * K + k0 + c1) * 2);
    ASYNC_LDS_B128(lds_lo32(&As[buf][r0 * 72 + c0]), ga0, A);
    ASYNC_LDS_B128(lds_lo32(&As[buf][r1 * 72 + c1]), ga1, A);
    ASYNC_LDS_B128(lds_lo32(&Bs[buf][r0 * 72 + c0]), gb0, Bt);
    ASYNC_LDS_B128(lds_lo32(&Bs[buf][r1 * 72 + c1]), gb1, Bt);
  };

  v8f acc0 = {}, acc1 = {};
  issue_tile(0, 0);
  WAIT_ASYNC();
  __syncthreads();
  int buf = 0;
  for (int k0 = 0; k0 < K; k0 += 64) {
    if (k0 + 64 < K) issue_tile(buf ^ 1, k0 + 64);   // overlap DMA with WMMA
#pragma unroll
    for (int kk = 0; kk < 64; kk += 32) {
      v16bf bfr = ld_b_bf16(Bs[buf], wc * 16, 72, kk, lane);
      v16bf a0 = ld_a_bf16(As[buf], wr * 32, 72, kk, lane);
      v16bf a1 = ld_a_bf16(As[buf], wr * 32 + 16, 72, kk, lane);
      acc0 = WMMA_BF16(a0, bfr, acc0);
      acc1 = WMMA_BF16(a1, bfr, acc1);
    }
    WAIT_ASYNC();
    __syncthreads();
    buf ^= 1;
  }

  const int n = bx * 64 + wc * 16 + (lane & 15);
  const int h8 = (lane >> 4) * 8;
#pragma unroll
  for (int t = 0; t < 2; ++t) {
    v8f acc = t ? acc1 : acc0;
#pragma unroll
    for (int r = 0; r < 8; ++r) {
      const int m = by * 64 + wr * 32 + t * 16 + h8 + r;
      const float v = acc[r];
      if (MODE == 0) {               // scatter into q|k|v, layout [B,H,T,D]
        const int which = n >> 9, hh = (n >> 6) & 7, d = n & 63;
        const int bb = m >> 9, tt = m & 511;
        outB[(size_t)which * (16384 * 64) +
             ((size_t)(bb * 8 + hh) * 512 + tt) * 64 + d] = (bf16_t)v;
      } else if (MODE == 1) {
        const size_t o = (size_t)m * N + n;
        outF[o] = v + bias[n] + resid[o];
      } else if (MODE == 2) {
        const size_t o = (size_t)m * N + n;
        const float z = v + bias[n];
        outB[o] = (bf16_t)(z > 0.f ? z : 0.f);
      } else {
        const size_t o = (size_t)m * N + n;
        outF[o] = v + bias[n] + resid[o];
      }
    }
  }
}

// ---------------------------------------------------------------------------
// Causal attention, one workgroup per (b, h, 64-query block). Exploits the
// 320KB WGP LDS: full fp32 score strip [64 x 512] stays resident, so softmax
// is a straight two-pass over LDS (no online rescaling). V is staged
// transposed so the PV WMMA B-fragments are contiguous along K(=s).
// Dynamic LDS = 219,136 B -> one workgroup per WGP. q/k tiles arrive via
// async global->LDS DMA.
// ---------------------------------------------------------------------------
__global__ __launch_bounds__(256) void attn_kernel(
    const bf16_t* __restrict__ q, const bf16_t* __restrict__ k,
    const bf16_t* __restrict__ v, bf16_t* __restrict__ o) {
  extern __shared__ __align__(16) char smem[];
  bf16_t* qs  = (bf16_t*)smem;                 // [64][72]
  bf16_t* ks  = qs + 64 * 72;                  // [64][72]
  bf16_t* vts = ks + 64 * 72;                  // [64 d][520 s]  (V transposed)
  float*  ss  = (float*)(vts + 64 * 520);      // [64][520] fp32 scores/probs
  float*  red = ss + 64 * 520;                 // [64][4] partials

  const int bid = blockIdx.x;                  // b*64 + h*8 + qb
  const int qb = bid & 7, hh = (bid >> 3) & 7, bb = bid >> 6;
  const int t0 = qb * 64;
  const size_t head = (size_t)(bb * 8 + hh) * 512 * 64;
  const bf16_t* qh = q + head;
  const bf16_t* kh = k + head;
  const bf16_t* vh = v + head;

  const int tid = threadIdx.x, lane = tid & 31, wave = tid >> 5;
  const int wr = wave >> 2, wc = wave & 3;

  // q block -> LDS via async DMA (once); completion covered by first in-loop
  // s_wait_asynccnt + barrier.
#pragma unroll
  for (int j = 0; j < 2; ++j) {
    const int e = tid * 2 + j;                 // 0..511 chunks of 8 bf16
    const int r = e >> 3, c8 = (e & 7) * 8;
    ASYNC_LDS_B128(lds_lo32(&qs[r * 72 + c8]),
                   (uint32_t)(((size_t)(t0 + r) * 64 + c8) * 2), qh);
  }

  const int nch = qb + 1;                      // causal: only chunks s <= t
  const float scale = 0.044194173824159216f;   // 512^-0.5 (reference uses C)

  // Phase 1: S = q k^T * scale, chunk by chunk, into fp32 LDS strip.
  for (int ch = 0; ch < nch; ++ch) {
    const int s0 = ch * 64;
#pragma unroll
    for (int j = 0; j < 2; ++j) {              // k chunk [64 s][64 d], async
      const int e = tid * 2 + j;
      const int r = e >> 3, c8 = (e & 7) * 8;
      ASYNC_LDS_B128(lds_lo32(&ks[r * 72 + c8]),
                     (uint32_t)(((size_t)(s0 + r) * 64 + c8) * 2), kh);
    }
#pragma unroll
    for (int j = 0; j < 16; ++j) {             // v chunk transposed -> vts
      const int e = tid * 16 + j;              // 0..4095
      const int sl = e >> 6, d = e & 63;
      vts[d * 520 + s0 + sl] = vh[(size_t)(s0 + sl) * 64 + d];
    }
    WAIT_ASYNC();
    __syncthreads();

    v8f s_acc0 = {}, s_acc1 = {};
#pragma unroll
    for (int kb = 0; kb < 64; kb += 32) {
      v16bf bf = ld_b_bf16(ks, wc * 16, 72, kb, lane);
      v16bf a0 = ld_a_bf16(qs, wr * 32, 72, kb, lane);
      v16bf a1 = ld_a_bf16(qs, wr * 32 + 16, 72, kb, lane);
      s_acc0 = WMMA_BF16(a0, bf, s_acc0);
      s_acc1 = WMMA_BF16(a1, bf, s_acc1);
    }
    const int n = wc * 16 + (lane & 15);
    const int h8 = (lane >> 4) * 8;
#pragma unroll
    for (int t = 0; t < 2; ++t) {
      v8f acc = t ? s_acc1 : s_acc0;
#pragma unroll
      for (int r = 0; r < 8; ++r)
        ss[(wr * 32 + t * 16 + h8 + r) * 520 + s0 + n] = acc[r] * scale;
    }
    __syncthreads();
  }

  // Phase 2: causal softmax over LDS. 4 threads per row, two-pass.
  const int kLen = nch * 64;
  const int row = tid >> 2, part = tid & 3;
  const int tg = t0 + row;                     // global query index
  const int span = kLen >> 2;                  // multiple of 16
  const int c0s = part * span;
  float* srow = ss + row * 520;

  float mx = -3.0e38f;
  for (int c = c0s; c < c0s + span; ++c)
    if (c <= tg) mx = fmaxf(mx, srow[c]);
  red[row * 4 + part] = mx;
  __syncthreads();
  const float rmax = fmaxf(fmaxf(red[row * 4], red[row * 4 + 1]),
                           fmaxf(red[row * 4 + 2], red[row * 4 + 3]));
  __syncthreads();
  float lsum = 0.f;
  for (int c = c0s; c < c0s + span; ++c)
    if (c <= tg) lsum += __expf(srow[c] - rmax);
  red[row * 4 + part] = lsum;
  __syncthreads();
  const float inv = 1.f / (red[row * 4] + red[row * 4 + 1] +
                           red[row * 4 + 2] + red[row * 4 + 3]);
  for (int c = c0s; c < c0s + span; ++c)
    srow[c] = (c <= tg) ? __expf(srow[c] - rmax) * inv : 0.f;
  __syncthreads();

  // Phase 3: O = P @ V  (A from fp32 probs, B from transposed V in LDS).
  v8f o0 = {}, o1 = {};
  for (int kb = 0; kb < kLen; kb += 32) {
    v16bf bf = ld_b_bf16(vts, wc * 16, 520, kb, lane);
    v16bf a0 = ld_a_f32(ss, wr * 32, 520, kb, lane);
    v16bf a1 = ld_a_f32(ss, wr * 32 + 16, 520, kb, lane);
    o0 = WMMA_BF16(a0, bf, o0);
    o1 = WMMA_BF16(a1, bf, o1);
  }
  const int n = wc * 16 + (lane & 15);         // d within head
  const int h8 = (lane >> 4) * 8;
#pragma unroll
  for (int t = 0; t < 2; ++t) {
    v8f acc = t ? o1 : o0;
#pragma unroll
    for (int r = 0; r < 8; ++r) {
      const int m = wr * 32 + t * 16 + h8 + r;
      o[((size_t)(bb * 512) + t0 + m) * 512 + hh * 64 + n] = (bf16_t)acc[r];
    }
  }
}

// ---------------------------------------------------------------------------
// Host orchestration.
// ---------------------------------------------------------------------------
extern "C" void kernel_launch(void* const* d_in, const int* in_sizes, int n_in,
                              void* d_out, int out_size, void* d_ws, size_t ws_size,
                              hipStream_t stream) {
  (void)in_sizes; (void)n_in; (void)out_size; (void)ws_size;
  constexpr int B_ = 32, T_ = 512, C_ = 512, D_ = 64, F_ = 2048;
  constexpr int R_ = B_ * T_;                  // 16384 token rows

  const float* x      = (const float*)d_in[0];
  const float* wq     = (const float*)d_in[1];
  const float* wk     = (const float*)d_in[2];
  const float* wv     = (const float*)d_in[3];
  const float* w_proj = (const float*)d_in[4];
  const float* b_proj = (const float*)d_in[5];
  const float* w1     = (const float*)d_in[6];
  const float* b1     = (const float*)d_in[7];
  const float* w2     = (const float*)d_in[8];
  const float* b2     = (const float*)d_in[9];
  const float* ln1g   = (const float*)d_in[10];
  const float* ln1b   = (const float*)d_in[11];
  const float* ln2g   = (const float*)d_in[12];
  const float* ln2b   = (const float*)d_in[13];

  char* ws = (char*)d_ws;
  size_t off = 0;
  auto alloc = [&](size_t bytes) {
    void* p = ws + off;
    off += (bytes + 255) & ~(size_t)255;
    return p;
  };
  bf16_t* hb    = (bf16_t*)alloc((size_t)R_ * C_ * 2);      // ln1 out
  bf16_t* wqkvT = (bf16_t*)alloc((size_t)3 * C_ * C_ * 2);  // [1536][512]
  bf16_t* qkv   = (bf16_t*)alloc((size_t)3 * R_ * D_ * 2);  // q|k|v [B,H,T,D]
  bf16_t* ob    = (bf16_t*)alloc((size_t)R_ * C_ * 2);      // attn out (concat)
  bf16_t* wpT   = (bf16_t*)alloc((size_t)C_ * C_ * 2);
  float*  x1    = (float*)alloc((size_t)R_ * C_ * 4);       // x + attn proj
  bf16_t* h2b   = (bf16_t*)alloc((size_t)R_ * C_ * 2);      // ln2 out
  bf16_t* w1T   = (bf16_t*)alloc((size_t)F_ * C_ * 2);
  bf16_t* w2T   = (bf16_t*)alloc((size_t)C_ * F_ * 2);
  bf16_t* a1    = (bf16_t*)alloc((size_t)R_ * F_ * 2);      // relu(h2 w1 + b1)

  bf16_t* qb_ = qkv;
  bf16_t* kb_ = qkv + (size_t)R_ * D_;
  bf16_t* vb_ = qkv + (size_t)2 * R_ * D_;

  // Weight prep (tiny; runs once per launch).
  qkvw_pack_kernel<<<(3 * C_ * C_ + 255) / 256, 256, 0, stream>>>(wq, wk, wv, wqkvT);
  transpose_bf16_kernel<<<(C_ * C_ + 255) / 256, 256, 0, stream>>>(w_proj, wpT, C_, C_);
  transpose_bf16_kernel<<<(C_ * F_ + 255) / 256, 256, 0, stream>>>(w1, w1T, C_, F_);
  transpose_bf16_kernel<<<(F_ * C_ + 255) / 256, 256, 0, stream>>>(w2, w2T, F_, C_);

  // h = LN1(x) -> bf16
  layernorm_bf16_kernel<<<R_, 256, 0, stream>>>(x, ln1g, ln1b, hb);

  // q,k,v = h @ Wqkv   (M=16384, N=1536, K=512), scatter to [B,H,T,D]
  gemm_bf16_kernel<0><<<dim3(1536 / 64, R_ / 64), 256, 0, stream>>>(
      hb, wqkvT, R_, 1536, C_, nullptr, nullptr, nullptr, qkv);

  // causal attention per (b, h, 64-row query block)
  constexpr size_t ATTN_LDS =
      (size_t)(64 * 72 + 64 * 72 + 64 * 520) * 2 + (size_t)(64 * 520 + 256) * 4;
  attn_kernel<<<B_ * 8 * (T_ / 64), 256, ATTN_LDS, stream>>>(qb_, kb_, vb_, ob);

  // x1 = x + o @ Wproj + b_proj   (fp32 residual path)
  gemm_bf16_kernel<1><<<dim3(C_ / 64, R_ / 64), 256, 0, stream>>>(
      ob, wpT, R_, C_, C_, b_proj, x, x1, nullptr);

  // h2 = LN2(x1) -> bf16
  layernorm_bf16_kernel<<<R_, 256, 0, stream>>>(x1, ln2g, ln2b, h2b);

  // a1 = relu(h2 @ W1 + b1) -> bf16   (M=16384, N=2048, K=512)
  gemm_bf16_kernel<2><<<dim3(F_ / 64, R_ / 64), 256, 0, stream>>>(
      h2b, w1T, R_, F_, C_, b1, nullptr, nullptr, a1);

  // out = x1 + a1 @ W2 + b2           (M=16384, N=512, K=2048)
  gemm_bf16_kernel<3><<<dim3(C_ / 64, R_ / 64), 256, 0, stream>>>(
      a1, w2T, R_, C_, F_, b2, x1, (float*)d_out, nullptr);
}
